// CapsuleLayer_18966575579387
// MI455X (gfx1250) — matrix operand order repaired and checked by reading, not compiled, not run on hardware
//
#include <hip/hip_runtime.h>
#include <hip/hip_bf16.h>

// ---------------------------------------------------------------------------
// CapsNet dynamic routing for MI455X (gfx1250, wave32, WMMA).
//   u_hat GEMM: (B*I=147456, J=128) x (J=128, K*D=1024) bf16 WMMA, f32 acc.
//               Epilogue drains C tiles LDS -> global via async-from-LDS b128
//               (128B-coalesced stores on the roofline-dominant 576MB write).
//   routing:    1 workgroup per batch, u_hat slice (4.7MB) stays L2-resident;
//               global_prefetch_b8 hides L2 latency in the reduce loops.
// ---------------------------------------------------------------------------

typedef __attribute__((ext_vector_type(16))) __bf16 v16bf;
typedef __attribute__((ext_vector_type(8)))  float  v8f;

#define BDIM 128
#define IDIM 1152
#define JDIM 128     // GEMM K dimension
#define NDIM 1024    // K*D = 32*32
#define MDIM (BDIM * IDIM)

#define TILE_M 128
#define TILE_N 64
#define LDA 136      // padded row stride (shorts): 272B rows, 16B aligned, 4-bank shift
#define LDB 136

union Frag16 { v16bf v; uint4 q[2]; };

__device__ __forceinline__ unsigned short f2bf(float f) {
  union { float f; unsigned int u; } x; x.f = f;
  unsigned int r = x.u + 0x7FFFu + ((x.u >> 16) & 1u);   // round-to-nearest-even
  return (unsigned short)(r >> 16);
}

// ---------------------------------------------------------------------------
// GEMM: U[147456][1024] = X[147456][128] * W[128][1024]  (bf16 WMMA, f32 acc)
// grid = (NDIM/TILE_N = 16, MDIM/TILE_M = 1152), block = 256 (8 waves)
// ---------------------------------------------------------------------------
__global__ __launch_bounds__(256, 2)
void caps_uhat_gemm(const float* __restrict__ X,
                    const float* __restrict__ W,
                    float* __restrict__ U) {
  __shared__ unsigned short lA[TILE_M * LDA];  // [m][k] bf16; reused as C stage
  __shared__ unsigned short lB[TILE_N * LDB];  // [n][k] bf16 (W tile transposed)

  const int tid    = threadIdx.x;
  const int nBlock = blockIdx.x;   // 0..15
  const int mBlock = blockIdx.y;   // 0..1151

  // ---- stage A tile: 128 rows x 128 K, fp32 -> bf16
  {
    const int r = tid >> 1;
    const int h = tid & 1;
    const float* src = X + (size_t)(mBlock * TILE_M + r) * JDIM + h * 64;
    unsigned short* dst = lA + r * LDA + h * 64;
#pragma unroll
    for (int j = 0; j < 16; ++j) {
      float4 f = *(const float4*)(src + 4 * j);
      unsigned int p0 = (unsigned int)f2bf(f.x) | ((unsigned int)f2bf(f.y) << 16);
      unsigned int p1 = (unsigned int)f2bf(f.z) | ((unsigned int)f2bf(f.w) << 16);
      *(uint2*)(dst + 4 * j) = make_uint2(p0, p1);
    }
  }
  // ---- stage B tile transposed: lB[n][j] = W[j][nBlock*64 + n]
  {
    const int j = tid >> 1;
    const int h = tid & 1;
    const float* src = W + (size_t)j * NDIM + nBlock * TILE_N + h * 32;
#pragma unroll
    for (int q = 0; q < 8; ++q) {
      float4 f = *(const float4*)(src + 4 * q);
      const int n = h * 32 + 4 * q;
      lB[(n + 0) * LDB + j] = f2bf(f.x);
      lB[(n + 1) * LDB + j] = f2bf(f.y);
      lB[(n + 2) * LDB + j] = f2bf(f.z);
      lB[(n + 3) * LDB + j] = f2bf(f.w);
    }
  }
  __syncthreads();

  const int lane   = tid & 31;
  const int wid    = tid >> 5;
  const int wm     = (wid & 3) * 32;   // wave M offset within tile (4 waves)
  const int wn     = (wid >> 2) * 32;  // wave N offset within tile (2 waves)
  const int lr     = lane & 15;
  const bool hiH   = lane >= 16;

  v8f acc[2][2] = {};

#pragma unroll
  for (int kk = 0; kk < JDIM; kk += 32) {
    Frag16 a[2], b[2];
    // A 16x32 bf16 frag: lanes0-15 row M=lr hold K kk+[0..7],[16..23];
    //                    lanes16-31 same rows hold K kk+[8..15],[24..31].
#pragma unroll
    for (int ms = 0; ms < 2; ++ms) {
      const unsigned short* p = lA + (wm + ms * 16 + lr) * LDA + kk + (hiH ? 8 : 0);
      a[ms].q[0] = *(const uint4*)(p);
      a[ms].q[1] = *(const uint4*)(p + 16);
    }
    // B 32x16 bf16 frag: lanes0-15 col N=lr hold K kk+[0..7],[16..23];
    //                    lanes16-31 hold K kk+[8..15],[24..31]. lB is [n][k].
#pragma unroll
    for (int ns = 0; ns < 2; ++ns) {
      const unsigned short* p = lB + (wn + ns * 16 + lr) * LDB + kk + (hiH ? 8 : 0);
      b[ns].q[0] = *(const uint4*)(p);
      b[ns].q[1] = *(const uint4*)(p + 16);
    }
#pragma unroll
    for (int ms = 0; ms < 2; ++ms)
#pragma unroll
      for (int ns = 0; ns < 2; ++ns)
        acc[ms][ns] = __builtin_amdgcn_wmma_f32_16x16x32_bf16(
            false, a[ms].v, false, b[ns].v, (short)0, acc[ms][ns], false, false);
  }

  // ---- epilogue: stage C 32x32 f32 per wave into LDS (reuse lA after barrier),
  //      then drain to global with 128B-coalesced async-from-LDS stores.
  __syncthreads();   // all waves done reading lA/lB fragments

  float* myC = (float*)lA + wid * 1024;   // 8 waves x 4KB = 32KB <= sizeof(lA)
#pragma unroll
  for (int ms = 0; ms < 2; ++ms)
#pragma unroll
    for (int ns = 0; ns < 2; ++ns)
#pragma unroll
      for (int v = 0; v < 8; ++v)
        myC[(ms * 16 + v + (hiH ? 8 : 0)) * 32 + ns * 16 + lr] = acc[ms][ns][v];

  // async store reads LDS through a path the compiler can't track: wait DS.
  asm volatile("s_wait_dscnt 0x0" ::: "memory");

  // flat low 32 bits == wave-relative LDS byte offset (LDS aperture rule)
  const unsigned ldsBase = (unsigned)(uintptr_t)myC;
  const size_t   gRow0   = (size_t)mBlock * TILE_M + wm;
  const size_t   gColB   = (size_t)(nBlock * TILE_N + wn) * sizeof(float);
  const char*    ubase   = (const char*)U;
#pragma unroll
  for (int q = 0; q < 8; ++q) {
    const int c   = q * 32 + lane;   // 256 x 16B chunks per wave tile
    const int row = c >> 3;          // 8 chunks per 128B row
    const int ch  = c & 7;
    uint64_t gaddr = (uint64_t)(uintptr_t)(ubase + (gRow0 + row) * (NDIM * 4) +
                                           gColB + (size_t)ch * 16);
    unsigned laddr = ldsBase + row * 128 + ch * 16;
    asm volatile("global_store_async_from_lds_b128 %0, %1, off"
                 :: "v"(gaddr), "v"(laddr) : "memory");
  }
  asm volatile("s_wait_asynccnt 0x0" ::: "memory");
}

// ---------------------------------------------------------------------------
// Dynamic routing: one workgroup per batch (128 blocks x 256 threads).
// Thread t owns (k,d) columns {t, t+256, t+512, t+768}; d == lane, so the
// squash / agreement reductions over d are single-wave __shfl_xor reductions.
// ---------------------------------------------------------------------------
__global__ __launch_bounds__(256, 1)
void caps_routing(const float* __restrict__ U,
                  float* __restrict__ bmat,
                  float* __restrict__ out) {
  __shared__ float rmax[IDIM];
  __shared__ float rsum[IDIM];
  __shared__ float vcap[1024];

  const int tid   = threadIdx.x;
  const int lane  = tid & 31;
  const int batch = blockIdx.x;
  const float* ub = U + (size_t)batch * IDIM * NDIM;
  float* bm       = bmat + (size_t)batch * IDIM * 32;

  // b logits start at zero
  for (int i = tid; i < IDIM * 32; i += 256) bm[i] = 0.f;
  __threadfence_block();
  __syncthreads();

  for (int it = 0; it < 3; ++it) {
    // phase A: per-row softmax stats over K=32
    for (int i = tid; i < IDIM; i += 256) {
      const float* row = bm + i * 32;
      float m = -3.4e38f;
#pragma unroll
      for (int k = 0; k < 32; ++k) m = fmaxf(m, row[k]);
      float s = 0.f;
#pragma unroll
      for (int k = 0; k < 32; ++k) s += __expf(row[k] - m);
      rmax[i] = m;
      rsum[i] = s;
    }
    __threadfence_block();
    __syncthreads();

    // phase B: s[k,d] = sum_i softmax(b)[i,k] * u_hat[i,k,d]
    float sacc[4] = {0.f, 0.f, 0.f, 0.f};
    for (int i = 0; i < IDIM; ++i) {
      const float rm  = rmax[i];
      const float inv = 1.0f / rsum[i];
      const float* urow = ub + (size_t)i * NDIM;
      if (i + 8 < IDIM) {  // cover L2 latency (-> global_prefetch_b8)
        const float* pf = urow + 8 * NDIM;
#pragma unroll
        for (int c = 0; c < 4; ++c) __builtin_prefetch(pf + tid + 256 * c, 0, 0);
      }
#pragma unroll
      for (int c = 0; c < 4; ++c) {
        const int kd = tid + 256 * c;
        const int k  = kd >> 5;
        const float cv = __expf(bm[i * 32 + k] - rm) * inv;
        sacc[c] = fmaf(cv, urow[kd], sacc[c]);
      }
    }

    // squash: per k, ss = sum_d s^2 (reduction over the 32 lanes == d)
#pragma unroll
    for (int c = 0; c < 4; ++c) {
      float sq = sacc[c] * sacc[c];
#pragma unroll
      for (int off = 16; off >= 1; off >>= 1) sq += __shfl_xor(sq, off, 32);
      const float scale = sq / ((1.0f + sq) * sqrtf(sq + 1e-7f));
      const float vv = scale * sacc[c];
      const int kd = tid + 256 * c;
      vcap[kd] = vv;
      if (it == 2) out[(size_t)batch * 1024 + kd] = vv;
    }
    __threadfence_block();
    __syncthreads();

    // phase C: b[i,k] += sum_d v[k,d] * u_hat[i,k,d]
    if (it < 2) {
      for (int i = 0; i < IDIM; ++i) {
        const float* urow = ub + (size_t)i * NDIM;
#pragma unroll
        for (int c = 0; c < 4; ++c) {
          const int kd = tid + 256 * c;
          float p = vcap[kd] * urow[kd];
#pragma unroll
          for (int off = 16; off >= 1; off >>= 1) p += __shfl_xor(p, off, 32);
          if (lane == 0) bm[i * 32 + (kd >> 5)] += p;  // unique (wave,c) -> k
        }
      }
      __threadfence_block();
      __syncthreads();
    }
  }
}

// ---------------------------------------------------------------------------
extern "C" void kernel_launch(void* const* d_in, const int* in_sizes, int n_in,
                              void* d_out, int out_size, void* d_ws, size_t ws_size,
                              hipStream_t stream) {
  const float* X = (const float*)d_in[0];   // [128, 1152, 128]
  const float* W = (const float*)d_in[1];   // [128, 32, 32] -> [128, 1024]
  float* out  = (float*)d_out;              // [128, 32, 32]
  float* uhat = (float*)d_ws;                                   // 576 MB
  float* bmat = uhat + (size_t)MDIM * NDIM;                     // + 18.9 MB

  caps_uhat_gemm<<<dim3(NDIM / TILE_N, MDIM / TILE_M), 256, 0, stream>>>(X, W, uhat);
  caps_routing<<<dim3(BDIM), 256, 0, stream>>>(uhat, bmat, out);
}
